// Pedal_65996467470545
// MI455X (gfx1250) — matrix operand order repaired and checked by reading, not compiled, not run on hardware
//
#include <hip/hip_runtime.h>

typedef __attribute__((ext_vector_type(16))) _Float16 v16h;
typedef __attribute__((ext_vector_type(8)))  float    v8f;

#define BATCH 32
#define HID   64
#define LDS_HSTRIDE 72                      // f16 elems per h-row; 144B, 16B-aligned
#define HBUF (BATCH * LDS_HSTRIDE)          // one h buffer (f16 elems)

union V16H {
  v16h v;
  float4 f4[2];
  _Float16 h[16];
};

union V8H {
  float4 f4;
  _Float16 h[8];
};

#if __has_builtin(__builtin_amdgcn_tanhf)
__device__ __forceinline__ float fast_tanh(float x) {
  return __builtin_amdgcn_tanhf(x);          // v_tanh_f32 (gfx1250 TRANS op)
}
#else
__device__ __forceinline__ float fast_tanh(float x) {
  return 1.0f - 2.0f * __builtin_amdgcn_rcpf(1.0f + __expf(2.0f * x));
}
#endif
__device__ __forceinline__ float fast_sigmoid(float x) {
  return 0.5f * fast_tanh(0.5f * x) + 0.5f;  // sigmoid via v_tanh
}

// One persistent workgroup: 8 waves of 32 on a single WGP.
// wave = (m = M-tile of 16 batch rows, cg = group of 16 hidden columns).
// Each wave computes all 4 gates for its (16 rows x 16 cols) patch per step.
// h is double-buffered in LDS -> single barrier per time step.
__global__ __launch_bounds__(256) void lstm_persistent(
    const float* __restrict__ x,      // [T, 32, 1]
    const float* __restrict__ W_ih,   // [256, 1]
    const float* __restrict__ W_hh,   // [256, 64] row-major
    const float* __restrict__ b_ih,   // [256]
    const float* __restrict__ b_hh,   // [256]
    const float* __restrict__ W_out,  // [1, 64]
    const float* __restrict__ b_out,  // [1]
    float* __restrict__ out,          // [T, 32, 1]
    int T)
{
  __shared__ __align__(16) _Float16 h_lds[2 * HBUF];

  const int tid    = threadIdx.x;
  const int wid    = tid >> 5;
  const int lane   = tid & 31;
  const int lane16 = lane & 15;
  const int hi     = (lane >> 4) & 1;     // which half of the wave
  const int m      = wid >> 2;            // M-tile: batch rows [m*16, m*16+16)
  const int cg     = wid & 3;             // hidden columns [cg*16, cg*16+16)
  const int hcol   = cg * 16 + lane16;    // this lane's hidden column (C/D layout: N = lane%16)

  // distributed output-projection assignment: wave w -> rows 4w..4w+3,
  // 8 lanes per row, lane j covers k = [8j, 8j+8)
  const int pb = wid * 4 + (lane >> 3);   // batch row this lane helps project
  const int pj = lane & 7;                // k-slice index

  // ---- init LDS: zero both h buffers ----
  for (int i = tid; i < 2 * HBUF; i += 256) h_lds[i] = (_Float16)0.0f;

  // ---- preload W_out slice (8 floats per lane, register resident) ----
  float wout_reg[8];
  #pragma unroll
  for (int q = 0; q < 8; ++q) wout_reg[q] = W_out[pj * 8 + q];

  // ---- preload W_hh as f16 B-matrices (resident in VGPRs) ----
  // B operand in this code's contiguous k order, same as A (32x16 KxN): lanes 0-15: N=lane, VGPR v holds K={2v,2v+1};
  // lanes 16-31: same N, K += 16.
  V16H Bw[4][2];
  float wih_c[4], bias_c[4];
  #pragma unroll
  for (int g = 0; g < 4; ++g) {
    const int gcol = g * 64 + cg * 16 + lane16;   // gate col (PyTorch i,f,g,o order)
    wih_c[g]  = W_ih[gcol];
    bias_c[g] = b_ih[gcol] + b_hh[gcol];
    #pragma unroll
    for (int s = 0; s < 2; ++s) {                 // K slice: [s*32, s*32+32)
      const int kbase = s * 32 + hi * 16;
      #pragma unroll
      for (int v = 0; v < 8; ++v) {
        const float* p = W_hh + (size_t)gcol * HID + kbase + 2 * v;
        Bw[g][s].h[2 * v]     = (_Float16)p[0];
        Bw[g][s].h[2 * v + 1] = (_Float16)p[1];
      }
    }
  }

  // c-state: 8 rows per lane (C/D layout: VGPR r -> row m*16 + r + 8*hi)
  float cst[8];
  #pragma unroll
  for (int r = 0; r < 8; ++r) cst[r] = 0.0f;

  const float bout = b_out[0];

  // pipelined x load: xl holds x[t, lane] at loop top
  float xl = x[lane];
  int ro = 0, wo = HBUF;   // read / write buffer offsets (swapped each step)

  __syncthreads();

  for (int t = 0; t < T; ++t) {
    const _Float16* __restrict__ hr = h_lds + ro;  // h_{t-1}
    _Float16* __restrict__ hw       = h_lds + wo;  // h_t

    // ---- load A operands (h_{t-1}, 16-bit A layout) via ds_read_b128 ----
    // lane<16: row=m*16+lane, K = s*32 + {0..15}
    // lane>=16: same row,     K = s*32 + {16..31}
    V16H A[2];
    {
      const int row = m * 16 + lane16;
      #pragma unroll
      for (int s = 0; s < 2; ++s) {
        const int o0 = row * LDS_HSTRIDE + s * 32 + hi * 16;
        A[s].f4[0] = *(const float4*)(&hr[o0]);
        A[s].f4[1] = *(const float4*)(&hr[o0 + 8]);
      }
    }

    // ---- distributed output projection for step t-1 (reads h_{t-1}) ----
    {
      V8H hh;
      hh.f4 = *(const float4*)(&hr[pb * LDS_HSTRIDE + pj * 8]);
      float p = 0.0f;
      #pragma unroll
      for (int q = 0; q < 8; ++q) p += (float)hh.h[q] * wout_reg[q];
      p += __shfl_xor(p, 1, 32);
      p += __shfl_xor(p, 2, 32);
      p += __shfl_xor(p, 4, 32);
      if (pj == 0 && t > 0)
        out[(size_t)(t - 1) * BATCH + pb] =
            p + bout + x[(size_t)(t - 1) * BATCH + pb];
    }

    // ---- gate accumulator init: xg = x*W_ih + b_ih + b_hh ----
    float xb[8];
    #pragma unroll
    for (int r = 0; r < 8; ++r)
      xb[r] = __shfl(xl, m * 16 + r + 8 * hi, 32);  // broadcast x for this lane's rows

    v8f acc[4];
    #pragma unroll
    for (int g = 0; g < 4; ++g) {
      v8f c;
      #pragma unroll
      for (int r = 0; r < 8; ++r) c[r] = xb[r] * wih_c[g] + bias_c[g];
      acc[g] = c;
    }

    // prefetch next step's x (clamped, branch-free) while WMMA runs
    const int tn = (t + 1 < T) ? (t + 1) : (T - 1);
    const float xl_next = x[(size_t)tn * BATCH + lane];

    // ---- gates += h_{t-1} @ W_hh^T : 8x v_wmma_f32_16x16x32_f16 ----
    #pragma unroll
    for (int s = 0; s < 2; ++s) {
      #pragma unroll
      for (int g = 0; g < 4; ++g) {
        acc[g] = __builtin_amdgcn_wmma_f32_16x16x32_f16(
            false, A[s].v, false, Bw[g][s].v, (short)0, acc[g], false, false);
      }
    }

    // ---- elementwise LSTM cell update; write h_t to the OTHER buffer ----
    #pragma unroll
    for (int r = 0; r < 8; ++r) {
      const float ig = fast_sigmoid(acc[0][r]);
      const float fg = fast_sigmoid(acc[1][r]);
      const float gg = fast_tanh(acc[2][r]);
      const float og = fast_sigmoid(acc[3][r]);
      const float c  = fg * cst[r] + ig * gg;
      cst[r] = c;
      const float h  = og * fast_tanh(c);
      const int row  = m * 16 + r + 8 * hi;
      hw[row * LDS_HSTRIDE + hcol] = (_Float16)h;
    }

    __syncthreads();   // h_t visible; also fences next step's writes vs this step's reads
    xl = xl_next;
    const int tmp = ro; ro = wo; wo = tmp;   // swap buffers
  }

  // ---- final output for t = T-1 (h_{T-1} now in buffer `ro`) ----
  {
    const _Float16* hr = h_lds + ro;
    V8H hh;
    hh.f4 = *(const float4*)(&hr[pb * LDS_HSTRIDE + pj * 8]);
    float p = 0.0f;
    #pragma unroll
    for (int q = 0; q < 8; ++q) p += (float)hh.h[q] * wout_reg[q];
    p += __shfl_xor(p, 1, 32);
    p += __shfl_xor(p, 2, 32);
    p += __shfl_xor(p, 4, 32);
    if (pj == 0)
      out[(size_t)(T - 1) * BATCH + pb] =
          p + bout + x[(size_t)(T - 1) * BATCH + pb];
  }
}

extern "C" void kernel_launch(void* const* d_in, const int* in_sizes, int n_in,
                              void* d_out, int out_size, void* d_ws, size_t ws_size,
                              hipStream_t stream) {
  const float* x     = (const float*)d_in[0];
  const float* W_ih  = (const float*)d_in[1];
  const float* W_hh  = (const float*)d_in[2];
  const float* b_ih  = (const float*)d_in[3];
  const float* b_hh  = (const float*)d_in[4];
  const float* W_out = (const float*)d_in[5];
  const float* b_out = (const float*)d_in[6];
  float* out = (float*)d_out;
  const int T = in_sizes[0] / BATCH;   // x is [T, 32, 1]

  lstm_persistent<<<1, 256, 0, stream>>>(x, W_ih, W_hh, b_ih, b_hh, W_out, b_out, out, T);
}
